// unit_gcgcn_32006096290303
// MI455X (gfx1250) — matrix operand order repaired and checked by reading, not compiled, not run on hardware
//
#include <hip/hip_runtime.h>

// unit_gcgcn for MI455X / gfx1250 (wave32, WMMA).
// Pipeline:
//   k_prep   : xm = mean_T(x); xh = f16 zero-padded copy of x  [b][c][8+T][32]
//   k_adj    : x1/x2 projections, tanh adjacency -> B-matrices (f16,
//              [s][b][k][i][v], v,i padded to 32), ridge output, W3 -> Ah f16
//   k_gemm_z : z = Sum_k,v xh * w  via v_wmma_f32_16x16x32_f16; k-outer /
//              M-tile-inner so only one B-fragment pair is live (no spills);
//              adjacency staged to LDS via async global->LDS when available;
//              fused (z-x)^2 loss partials; z stored f16 as
//              zt[b][t*25+v][s*64+c]  (kappa-contiguous for the second GEMM)
//   k_gemm_y : y = W3 . z  (M=64,K=192,N=6400 per batch) via WMMA, fused
//              b3-sum + BatchNorm + residual + ReLU into d_out
//   k_loss   : deterministic reduction of loss partials
//
// Workspace layout (bytes), total ~115.3 MB:
//   xh  f16 @ 0          : 32*64*264*32*2 = 34,603,008
//   xm  f32 @ 34,603,008 : 204,800
//   btg f16 @ 34,807,808 : 3*32*9*32*32*2 = 1,769,472
//   ah  f16 @ 36,577,280 : 64*192*2 = 24,576
//   lp  f32 @ 36,601,856 : 3072*4 = 12,288
//   zt  f16 @ 36,614,144 : 32*6400*192*2 = 78,643,200

#define BDIM 32
#define CDIM 64
#define OUTC 64
#define TDIM 256
#define VDIM 25
#define SDIM 3
#define TS   9

#define LOSS_OFF  13107200        // y elems, then loss (3), then ridge (96)
#define RIDGE_OFF 13107203

#if __has_builtin(__builtin_amdgcn_global_load_async_to_lds_b128) && \
    __has_builtin(__builtin_amdgcn_s_wait_asynccnt)
#define HAVE_ASYNC_LDS 1
#endif

typedef _Float16 h8   __attribute__((ext_vector_type(8)));
typedef _Float16 v16h __attribute__((ext_vector_type(16)));
typedef float    v8f  __attribute__((ext_vector_type(8)));
typedef int      i4   __attribute__((ext_vector_type(4)));

__device__ __forceinline__ v16h pack16(h8 lo, h8 hi) {
  return __builtin_shufflevector(lo, hi, 0,1,2,3,4,5,6,7,8,9,10,11,12,13,14,15);
}
__device__ __forceinline__ v8f wmma16(v16h a, v16h b, v8f c) {
  // D = A(16x32 f16) * B(32x16 f16) + C(16x16 f32)
  return __builtin_amdgcn_wmma_f32_16x16x32_f16(false, a, false, b, (short)0, c,
                                                false, false);
}

#ifdef HAVE_ASYNC_LDS
__device__ __forceinline__ void async_g2l_b128(const _Float16* g, _Float16* l) {
  __builtin_amdgcn_global_load_async_to_lds_b128(
      (__attribute__((address_space(1))) i4*)(_Float16*)g,
      (__attribute__((address_space(3))) i4*)l, 0, 0);
}
#endif

// ---------------------------------------------------------------------------
// k_prep: mean over T and padded f16 copy of x.
// grid = B*C blocks, 256 threads (8 rows x 32 lanes per iteration).
__global__ void k_prep(const float* __restrict__ x, _Float16* __restrict__ xh,
                       float* __restrict__ xm) {
  const int bc  = blockIdx.x;          // b*64 + c
  const int tid = threadIdx.x;
  const int vp  = tid & 31, ty = tid >> 5;
  __shared__ float part[8][32];
  _Float16* xrow = xh + (size_t)bc * (264 * 32);
  xrow[ty * 32 + vp] = (_Float16)0.f;  // causal zero pad rows u = 0..7
  const float* xb = x + (size_t)bc * (TDIM * VDIM);
  float ms = 0.f;
  for (int it = 0; it < 32; ++it) {
    const int t = it * 8 + ty;
    float val = 0.f;
    if (vp < VDIM) val = xb[t * VDIM + vp];
    xrow[(t + 8) * 32 + vp] = (_Float16)val;   // v padded to 32 with zeros
    ms += val;
  }
  part[ty][vp] = ms;
  __syncthreads();
  if (ty == 0 && vp < VDIM) {
    float s = 0.f;
    for (int r = 0; r < 8; ++r) s += part[r][vp];
    xm[bc * VDIM + vp] = s * (1.f / TDIM);
  }
}

// ---------------------------------------------------------------------------
// k_adj: x1/x2, tanh adjacency -> btg[s][b][k][i][v] (f16, 32x32 padded),
// ridge output, plus Ah = W3 rearranged [o][kappa=s*64+c] (block 0 only).
// grid = S*B blocks, 256 threads.
__global__ void k_adj(const float* __restrict__ xm, const float* __restrict__ Amat,
                      const float* __restrict__ alpha,
                      const float* __restrict__ W1, const float* __restrict__ b1,
                      const float* __restrict__ W2, const float* __restrict__ b2,
                      const float* __restrict__ W3,
                      _Float16* __restrict__ btg, _Float16* __restrict__ ah,
                      float* __restrict__ out) {
  const int sb = blockIdx.x, s = sb / BDIM, b = sb % BDIM;
  const int tid = threadIdx.x;
  __shared__ float x1s[TS * VDIM], x2s[TS * VDIM];
  __shared__ float rr[256];
  if (tid < 2 * TS * VDIM) {                 // 450 threads-worth of tiny dots
    const bool first = tid < TS * VDIM;
    const int idx = first ? tid : tid - TS * VDIM;
    const int k = idx / VDIM, v = idx % VDIM;
    const float* Wp = (first ? W1 : W2) + (s * TS + k) * CDIM;
    float acc = (first ? b1 : b2)[s * TS + k];
    const float* xmb = xm + b * CDIM * VDIM + v;
    for (int c = 0; c < CDIM; ++c) acc += xmb[c * VDIM] * Wp[c];
    (first ? x1s : x2s)[idx] = acc;
  }
  __syncthreads();
  const float alpha0 = alpha[0];
  float rsum = 0.f;
  _Float16* bdst = btg + (size_t)sb * (TS * 32 * 32);
  for (int e = tid; e < TS * 32 * 32; e += 256) {
    const int k = e >> 10, r = e & 1023, i = r >> 5, v = r & 31;
    float val = 0.f;
    if (i < VDIM && v < VDIM) {
      const float th = tanhf(x1s[k * VDIM + v] - x2s[k * VDIM + i]);
      rsum += th * th;
      val = alpha0 * th + Amat[(s * VDIM + v) * VDIM + i];
    }
    bdst[e] = (_Float16)val;                 // [k][i][v], padded with zeros
  }
  rr[tid] = rsum;
  __syncthreads();
  for (int off = 128; off > 0; off >>= 1) {
    if (tid < off) rr[tid] += rr[tid + off];
    __syncthreads();
  }
  if (tid == 0) out[RIDGE_OFF + sb] = rr[0];
  if (sb == 0) {                             // Ah[o][kappa] = W3[s2,o,c]
    for (int e = tid; e < OUTC * 192; e += 256) {
      const int o = e / 192, kk = e % 192, s2 = kk >> 6, c = kk & 63;
      ah[e] = (_Float16)W3[(s2 * OUTC + o) * CDIM + c];
    }
  }
}

// ---------------------------------------------------------------------------
// k_gemm_z: per (s,b): z(16384x25) = A(16384x288) * B(288x32), K' = 9 taps x 32.
// grid = S*B*32 blocks, 256 threads = 8 waves; each wave holds 4 M-tiles of
// accumulators (64 VGPRs) and streams taps k-outer so only one B-fragment
// pair (16 VGPRs) is live at a time -> no scratch spills.
__global__ void k_gemm_z(const _Float16* __restrict__ xh,
                         const _Float16* __restrict__ btg,
                         const float* __restrict__ x,
                         _Float16* __restrict__ zt,
                         float* __restrict__ lossPart) {
  const int sb = blockIdx.x >> 5, chunk = blockIdx.x & 31;
  const int s = sb / BDIM, b = sb % BDIM;
  const int tid = threadIdx.x, lane = tid & 31, w = tid >> 5;
  const int tp = lane & 15, hi = lane >> 4;
  __shared__ _Float16 Bsm[TS * 32 * 32];     // 18 KB adjacency for this (s,b)
  __shared__ float red[256];

  const _Float16* gB = btg + (size_t)sb * (TS * 32 * 32);
#ifdef HAVE_ASYNC_LDS
  for (int i = tid; i < (TS * 32 * 32) / 8; i += 256)
    async_g2l_b128(gB + i * 8, &Bsm[i * 8]);
  __builtin_amdgcn_s_wait_asynccnt(0);
#else
  {
    const h8* src = (const h8*)gB;
    h8* dst = (h8*)Bsm;
    for (int i = tid; i < (TS * 32 * 32) / 8; i += 256) dst[i] = src[i];
  }
#endif
  __syncthreads();

  const int mt0 = chunk * 32 + w * 4;        // this wave's 4 M-tiles
  const _Float16* abase[4];
#pragma unroll
  for (int it = 0; it < 4; ++it) {
    const int mt = mt0 + it;
    abase[it] = xh + ((size_t)(b * CDIM + (mt >> 4)) * 264 +
                      ((mt & 15) << 4) + tp) * 32 + hi * 8;
    __builtin_prefetch(abase[it], 0, 1);     // pull A slab toward the WGP
  }

  v8f acc[4][2] = {};
#pragma unroll
  for (int k = 0; k < TS; ++k) {
    // B fragments for tap k: column i = tp (+16), v = hi*8+[0,8) and 16+hi*8+[0,8)
    const _Float16* bp0 = &Bsm[k * 1024 + tp * 32 + hi * 8];
    const v16h b0 = pack16(*(const h8*)bp0, *(const h8*)(bp0 + 16));
    const _Float16* bp1 = &Bsm[k * 1024 + (tp + 16) * 32 + hi * 8];
    const v16h b1f = pack16(*(const h8*)bp1, *(const h8*)(bp1 + 16));
#pragma unroll
    for (int it = 0; it < 4; ++it) {
      const _Float16* ap = abase[it] + k * 32;   // A row t'=tp, shifted by tap
      const v16h a = pack16(*(const h8*)ap, *(const h8*)(ap + 16));
      acc[it][0] = wmma16(a, b0, acc[it][0]);
      acc[it][1] = wmma16(a, b1f, acc[it][1]);
    }
  }

  // Epilogue: D layout vgpr j, lane -> (t = tbase + j + hi*8, i = tp [+16])
  float lsum = 0.f;
  _Float16* ztb = zt + (size_t)b * (6400 * 192) + s * 64;   // kappa base s*64
#pragma unroll
  for (int it = 0; it < 4; ++it) {
    const int mt = mt0 + it;
    const int c = mt >> 4;
    const int tbase = (mt & 15) << 4;
    const float* xcb = x + (size_t)(b * CDIM + c) * TDIM * VDIM;
#pragma unroll
    for (int j = 0; j < 8; ++j) {
      const int t = tbase + j + hi * 8;
      const float z0 = acc[it][0][j];
      const float d0 = z0 - xcb[t * VDIM + tp];
      lsum += d0 * d0;
      ztb[(size_t)(t * VDIM + tp) * 192 + c] = (_Float16)z0;
      const int i1 = tp + 16;
      if (i1 < VDIM) {
        const float z1 = acc[it][1][j];
        const float d1 = z1 - xcb[t * VDIM + i1];
        lsum += d1 * d1;
        ztb[(size_t)(t * VDIM + i1) * 192 + c] = (_Float16)z1;
      }
    }
  }
  red[tid] = lsum;
  __syncthreads();
  for (int off = 128; off > 0; off >>= 1) {
    if (tid < off) red[tid] += red[tid + off];
    __syncthreads();
  }
  if (tid == 0) lossPart[blockIdx.x] = red[0];
}

// ---------------------------------------------------------------------------
// k_gemm_y: per b: y(64 x 6400) = Ah(64x192) * zt_b(192x6400), fused
// b3-sum + BatchNorm(eval) + residual + ReLU. grid = B*50, 256 threads.
__global__ void k_gemm_y(const _Float16* __restrict__ zt,
                         const _Float16* __restrict__ ah,
                         const float* __restrict__ x, const float* __restrict__ b3,
                         const float* __restrict__ bg, const float* __restrict__ bb,
                         const float* __restrict__ bm, const float* __restrict__ bv,
                         float* __restrict__ out) {
  const int b = blockIdx.x / 50, chunk = blockIdx.x % 50;
  const int tid = threadIdx.x, lane = tid & 31, w = tid >> 5;
  const int tp = lane & 15, hi = lane >> 4;
  const int ntile = chunk * 8 + w;           // 0..399
  const int n = ntile * 16 + tp;             // column = t*25 + v
  v8f acc[4] = {};
  const _Float16* ztb = zt + (size_t)b * (6400 * 192);
#pragma unroll
  for (int kt = 0; kt < 6; ++kt) {
    const _Float16* bp = ztb + (size_t)n * 192 + kt * 32 + hi * 8;
    const v16h bf = pack16(*(const h8*)bp, *(const h8*)(bp + 16));
#pragma unroll
    for (int mt = 0; mt < 4; ++mt) {
      const _Float16* ap = ah + (size_t)(mt * 16 + tp) * 192 + kt * 32 + hi * 8;
      const v16h af = pack16(*(const h8*)ap, *(const h8*)(ap + 16));
      acc[mt] = wmma16(af, bf, acc[mt]);
    }
  }
  const int t = n / VDIM, v = n - t * VDIM;
#pragma unroll
  for (int mt = 0; mt < 4; ++mt) {
#pragma unroll
    for (int j = 0; j < 8; ++j) {
      const int o = mt * 16 + j + hi * 8;
      float yv = acc[mt][j] + b3[o] + b3[OUTC + o] + b3[2 * OUTC + o];
      const float inv = bg[o] * rsqrtf(bv[o] + 1e-5f);
      yv = (yv - bm[o]) * inv + bb[o];
      const size_t xi = ((size_t)(b * OUTC + o) * TDIM + t) * VDIM + v;
      yv += x[xi];                            // identity residual
      out[xi] = fmaxf(yv, 0.f);
    }
  }
}

// ---------------------------------------------------------------------------
// k_loss: deterministic reduction of 1024 partials per subset.
__global__ void k_loss(const float* __restrict__ lossPart, float* __restrict__ out) {
  const int s = blockIdx.x, tid = threadIdx.x;
  __shared__ float sm[256];
  float a = 0.f;
  for (int i = tid; i < 1024; i += 256) a += lossPart[s * 1024 + i];
  sm[tid] = a;
  __syncthreads();
  for (int off = 128; off > 0; off >>= 1) {
    if (tid < off) sm[tid] += sm[tid + off];
    __syncthreads();
  }
  if (tid == 0) out[LOSS_OFF + s] = sm[0] * (1.f / 13107200.f);
}

// ---------------------------------------------------------------------------
extern "C" void kernel_launch(void* const* d_in, const int* in_sizes, int n_in,
                              void* d_out, int out_size, void* d_ws, size_t ws_size,
                              hipStream_t stream) {
  const float* x     = (const float*)d_in[0];
  const float* A     = (const float*)d_in[1];
  const float* alpha = (const float*)d_in[2];
  const float* W1    = (const float*)d_in[3];
  const float* b1    = (const float*)d_in[4];
  const float* W2    = (const float*)d_in[5];
  const float* b2    = (const float*)d_in[6];
  const float* W3    = (const float*)d_in[7];
  const float* b3    = (const float*)d_in[8];
  const float* bg    = (const float*)d_in[9];
  const float* bb    = (const float*)d_in[10];
  const float* bm    = (const float*)d_in[11];
  const float* bv    = (const float*)d_in[12];
  float* out = (float*)d_out;
  char* ws = (char*)d_ws;

  _Float16* xh  = (_Float16*)(ws + 0);
  float*    xm  = (float*)(ws + 34603008);
  _Float16* btg = (_Float16*)(ws + 34807808);
  _Float16* ah  = (_Float16*)(ws + 36577280);
  float*    lp  = (float*)(ws + 36601856);
  _Float16* zt  = (_Float16*)(ws + 36614144);

  k_prep<<<BDIM * CDIM, 256, 0, stream>>>(x, xh, xm);
  k_adj<<<SDIM * BDIM, 256, 0, stream>>>(xm, A, alpha, W1, b1, W2, b2, W3,
                                         btg, ah, out);
  k_gemm_z<<<SDIM * BDIM * 32, 256, 0, stream>>>(xh, btg, x, zt, lp);
  k_gemm_y<<<BDIM * 50, 256, 0, stream>>>(zt, ah, x, b3, bg, bb, bm, bv, out);
  k_loss<<<SDIM, 256, 0, stream>>>(lp, out);

  (void)in_sizes; (void)n_in; (void)out_size; (void)ws_size;
}
